// coRNN_12025908429051
// MI455X (gfx1250) — matrix operand order repaired and checked by reading, not compiled, not run on hardware
//
#include <hip/hip_runtime.h>
#include <hip/hip_bf16.h>

// ---------------------------------------------------------------------------
// coRNN on MI455X (gfx1250): fused-recurrence persistent kernel.
//   pre = tanh([hz | hy | x_t] @ [Wz | Wy | Wi].T + bi)   (K = 1152, bf16 WMMA)
//   hz += DT*(pre - GAMMA*hy - EPSILON*hz);  hy += DT*hz  (f32 in registers)
// 32 persistent WGs x 256 threads; weights slice async-staged into LDS once;
// state double-buffered in L2 as bf16; agent-scope atomic barrier per step.
// ---------------------------------------------------------------------------

typedef __bf16 bf16;
typedef __attribute__((ext_vector_type(16))) __bf16 v16bf;
typedef __attribute__((ext_vector_type(8)))  __bf16 v8bf;
typedef __attribute__((ext_vector_type(8)))  float  v8f;
typedef int v4i __attribute__((vector_size(16)));   // async-to-LDS builtin operand

#define T_STEPS 1024
#define BATCH   128
#define NINP    128
#define NHID    512
#define K_TOT   (2*NHID + NINP)     // 1152
#define NWG     32
#define COLS_PER_WG 16              // 32 WGs * 16 cols = 512 hidden units
#define DT_C      0.042f
#define GAMMA_C   2.7f
#define EPS_C     4.7f

union Frag { v16bf v; v8bf h[2]; };

// Branch-free tanh on the serial critical path: native v_tanh_f32 (confirmed
// in round-2 disasm), with a v_exp_f32 rational fallback (no EXEC churn).
__device__ __forceinline__ float fast_tanh(float z) {
#if __has_builtin(__builtin_amdgcn_tanhf)
    return __builtin_amdgcn_tanhf(z);
#else
    float e = __expf(-2.0f * fabsf(z));     // e in (0,1], never overflows
    float t = (1.0f - e) / (1.0f + e);
    return copysignf(t, z);
#endif
}

// --- prep: build bf16 [Wz|Wy|Wi] (512x1152), zero state bufs, seed x_0, reset barrier
__global__ __launch_bounds__(256) void corn_prep(
    const float* __restrict__ x, const float* __restrict__ Wi,
    const float* __restrict__ Wh, bf16* __restrict__ Wcat,
    bf16* __restrict__ A0, bf16* __restrict__ A1, unsigned* __restrict__ counter)
{
    const int NW = NHID * K_TOT;            // 589824 weight elems
    const int NZ = 2 * BATCH * 2 * NHID;    // 262144 state elems (both buffers)
    int idx = blockIdx.x * 256 + threadIdx.x;
    if (idx == 0) *counter = 0u;
    if (idx < NW) {
        int h = idx / K_TOT, k = idx % K_TOT;
        float w = (k < 2*NHID) ? Wh[h*(2*NHID) + k] : Wi[h*NINP + (k - 2*NHID)];
        Wcat[idx] = (bf16)w;
    } else if (idx < NW + NZ) {
        int r = idx - NW;
        int buf = r / (BATCH * 2*NHID);
        int q   = r % (BATCH * 2*NHID);
        int m = q / (2*NHID), k = q % (2*NHID);
        bf16* A = buf ? A1 : A0;
        A[(size_t)m * K_TOT + k] = (bf16)0.0f;
    } else if (idx < NW + NZ + BATCH*NINP) {
        int r = idx - NW - NZ;
        int b = r / NINP, i = r % NINP;
        A0[(size_t)b * K_TOT + 2*NHID + i] = (bf16)x[b*NINP + i];  // x[t=0]
    }
}

// --- main persistent recurrence kernel
__global__ __launch_bounds__(256) void corn_recurrence(
    const float* __restrict__ x, const float* __restrict__ bi,
    const bf16* __restrict__ Wcat, bf16* __restrict__ A0, bf16* __restrict__ A1,
    unsigned* __restrict__ counter, float* __restrict__ out)
{
    __shared__ __align__(16) bf16 sW[COLS_PER_WG * K_TOT];   // 36 KB weight slice

    const int g    = blockIdx.x;          // column-slice owner (0..31)
    const int tid  = threadIdx.x;
    const int w    = tid >> 5;            // wave id 0..7 == row tile
    const int lane = tid & 31;
    const int half = lane >> 4;
    const int l15  = lane & 15;

    // Stage this WG's 16 weight rows (contiguous 36864 B) into LDS once,
    // via the gfx1250 async copy path (ASYNCcnt) when available.
    {
        const bf16* src = Wcat + (size_t)g * COLS_PER_WG * K_TOT;
        const int n16 = COLS_PER_WG * K_TOT * (int)sizeof(bf16) / 16;   // 2304
#if __has_builtin(__builtin_amdgcn_global_load_async_to_lds_b128) && \
    __has_builtin(__builtin_amdgcn_s_wait_asynccnt)
        for (int i = tid; i < n16; i += 256)
            __builtin_amdgcn_global_load_async_to_lds_b128(
                (v4i*)(src + i * 8), (v4i*)(sW + i * 8), 0, 0);
        __builtin_amdgcn_s_wait_asynccnt(0);
#else
        const uint4* s4 = (const uint4*)src;
        uint4* d4 = (uint4*)sW;
        for (int i = tid; i < n16; i += 256) d4[i] = s4[i];
#endif
    }
    __syncthreads();

    const float  bias = bi[g * COLS_PER_WG + l15];
    const int    hcol = g * COLS_PER_WG + l15;
    const bf16*  srow = sW + (size_t)l15 * K_TOT;            // B operand row (LDS)
    const size_t arow_off = (size_t)(w * 16 + l15) * K_TOT;  // A operand row (global)
    const size_t crow_off = (size_t)(w * 16 + half * 8) * K_TOT + hcol; // C/D row 0

    v8f hz = {}, hy = {};   // f32 ODE state lives in registers for all 1024 steps

    for (int t = 0; t < T_STEPS; ++t) {
        const bf16* cur = (t & 1) ? A1 : A0;
        bf16*       nxt = (t & 1) ? A0 : A1;
        const bf16* arow = cur + arow_off;
        const bool  last = (t == T_STEPS - 1);

        // Early, independent work: convert x_{t+1} slice (rows 4g..4g+3) into
        // the next A buffer; overlaps with the WMMA load clauses below.
        if (!last) {
            const float* xs = x + (size_t)(t + 1) * BATCH * NINP;
            const int e = tid * 2;
            const int b = 4 * g + (e >> 7);
            const int i = e & 127;
            float2 xv = *(const float2*)(xs + b * NINP + i);
            bf16* dp = nxt + (size_t)b * K_TOT + 2*NHID + i;
            dp[0] = (bf16)xv.x; dp[1] = (bf16)xv.y;
        }

        // Two interleaved accumulators halve the dependent WMMA D->C chain.
        v8f acc0 = {}, acc1 = {};
        #pragma unroll 6
        for (int kc = 0; kc < K_TOT / 32; ++kc) {
            // A (16-bit, 16x32): lane holds K = {8*half..+8, 16+8*half..+8}
            const int ka = kc * 32 + half * 8;
            // B (16-bit, 32x16): lane holds K = {8*half..+8, 16+8*half..+8}
            const int kb = kc * 32 + half * 8;
            Frag a, b;
            a.h[0] = *(const v8bf*)(arow + ka);
            a.h[1] = *(const v8bf*)(arow + ka + 16);
            b.h[0] = *(const v8bf*)(srow + kb);
            b.h[1] = *(const v8bf*)(srow + kb + 16);
            if ((kc % 6) == 0)                       // near-scope, 512B ahead
                __builtin_prefetch(arow + ka + 256, 0, 3);
            if ((kc & 1) == 0)
                acc0 = __builtin_amdgcn_wmma_f32_16x16x32_bf16(
                           false, a.v, false, b.v, (short)0, acc0, false, false);
            else
                acc1 = __builtin_amdgcn_wmma_f32_16x16x32_bf16(
                           false, a.v, false, b.v, (short)0, acc1, false, false);
        }

        // State update in f32 registers (C/D layout: v -> M = 16w + v + 8*half).
        #pragma unroll
        for (int v = 0; v < 8; ++v) {
            float pre = fast_tanh((acc0[v] + acc1[v]) + bias);
            float nz  = hz[v] + DT_C * (pre - GAMMA_C * hy[v] - EPS_C * hz[v]);
            float ny  = hy[v] + DT_C * nz;
            hz[v] = nz; hy[v] = ny;
        }

        if (!last) {
            // Publish bf16 state; v*K_TOT folds into the store's imm offset.
            bf16* prow = nxt + crow_off;
            #pragma unroll
            for (int v = 0; v < 8; ++v) {
                prow[v * K_TOT]        = (bf16)hz[v];   // hz region (K = hcol)
                prow[v * K_TOT + NHID] = (bf16)hy[v];   // hy region (K = 512+hcol)
            }

            // Device-wide phase barrier (all 32 WGs resident).
            __syncthreads();
            if (tid == 0) {
                __hip_atomic_fetch_add(counter, 1u, __ATOMIC_RELEASE,
                                       __HIP_MEMORY_SCOPE_AGENT);
                const unsigned target = (unsigned)(t + 1) * NWG;
                while (__hip_atomic_load(counter, __ATOMIC_ACQUIRE,
                                         __HIP_MEMORY_SCOPE_AGENT) < target)
                    __builtin_amdgcn_s_sleep(1);
            }
            __syncthreads();
        }
    }

    // Emit final hy (f32, 128x512).
    #pragma unroll
    for (int v = 0; v < 8; ++v) {
        const int M = w * 16 + v + half * 8;
        out[(size_t)M * NHID + g * COLS_PER_WG + l15] = hy[v];
    }
}

extern "C" void kernel_launch(void* const* d_in, const int* in_sizes, int n_in,
                              void* d_out, int out_size, void* d_ws, size_t ws_size,
                              hipStream_t stream)
{
    (void)in_sizes; (void)n_in; (void)out_size; (void)ws_size;
    const float* x  = (const float*)d_in[0];   // (1024,128,128)
    const float* Wi = (const float*)d_in[1];   // (512,128)
    const float* bi = (const float*)d_in[2];   // (512,)
    const float* Wh = (const float*)d_in[3];   // (512,1024)
    float* out = (float*)d_out;                // (128,512)

    char* ws = (char*)d_ws;
    size_t off = 0;
    unsigned* counter = (unsigned*)(ws + off);          off += 256;
    bf16* Wcat = (bf16*)(ws + off);                     off += (size_t)NHID * K_TOT * sizeof(bf16);
    off = (off + 255) & ~(size_t)255;
    bf16* Abuf0 = (bf16*)(ws + off);                    off += (size_t)BATCH * K_TOT * sizeof(bf16);
    off = (off + 255) & ~(size_t)255;
    bf16* Abuf1 = (bf16*)(ws + off);
    // total ws use ~1.8 MB

    const int total = NHID * K_TOT + 2 * BATCH * 2 * NHID + BATCH * NINP;  // 868352
    corn_prep<<<(total + 255) / 256, 256, 0, stream>>>(x, Wi, Wh, Wcat, Abuf0, Abuf1, counter);
    corn_recurrence<<<NWG, 256, 0, stream>>>(x, bi, Wcat, Abuf0, Abuf1, counter, out);
}
